// UsageMemory_26972394619138
// MI455X (gfx1250) — compile-verified, matrix operand order checked by reading, not run on hardware
//
#include <hip/hip_runtime.h>
#include <hip/hip_bf16.h>

#define B_   256
#define M_   2048
#define E_   128
#define IN_  512
#define Q_   512
#define OUT_ 512
#define H_   64

typedef __attribute__((ext_vector_type(16))) __bf16 v16bf;
typedef __attribute__((ext_vector_type(8)))  float  v8f;

// ---------------------------------------------------------------------------
// CDNA5 async global->LDS copy (ASYNCcnt-tracked, no VGPR staging).
// lds_off = byte offset within the workgroup LDS allocation (low 32 bits of
// the generic pointer to a __shared__ object). gptr = 64-bit global address.
// ---------------------------------------------------------------------------
__device__ __forceinline__ void async_copy_b128(unsigned lds_off, const float* gptr) {
    asm volatile("global_load_async_to_lds_b128 %0, %1, off"
                 :: "v"(lds_off), "v"(gptr)
                 : "memory");
}
__device__ __forceinline__ void wait_async_all() {
    asm volatile("s_wait_asynccnt 0" ::: "memory");
}
__device__ __forceinline__ unsigned lds_off_of(const void* p) {
    return (unsigned)(unsigned long long)p;   // low 32 bits = LDS offset
}

// ---------------------------------------------------------------------------
// WMMA fragment gather from f32 LDS (row-major, row stride 128), converting
// to bf16 at gather time.
// 16-bit A-matrix 16x32 layout (ISA 7.12.2): lanes 0-15 hold K=0..7,16..23,
// lanes 16-31 hold K=8..15,24..31, row = lane%16.
// ---------------------------------------------------------------------------
__device__ __forceinline__ v16bf gather_frag(const float* lds, int row0, int k0, int lane) {
    int r    = row0 + (lane & 15);
    int half = lane >> 4;
    const float* p = lds + r * E_ + k0 + half * 8;
    v16bf v;
#pragma unroll
    for (int i = 0; i < 8; ++i) v[i] = (__bf16)p[i];
#pragma unroll
    for (int i = 0; i < 8; ++i) v[8 + i] = (__bf16)p[16 + i];
    return v;
}

// ---------------------------------------------------------------------------
// One block = 8 waves computing a 32(m) x 64(h) tile of A(32x128) @ B(64x128)^T
// in bf16 WMMA, then reducing against vec[h] (+bias[h]) into scoreAcc[32].
// C layout: VGPR r, lanes 0-15 -> M=r, lanes 16-31 -> M=8+r; N = lane%16.
// ---------------------------------------------------------------------------
__device__ __forceinline__ void wmma_tile_scores(const float* ldsA, const float* ldsB,
                                                 const float* vec, const float* bias,
                                                 float* scoreAcc, int tid) {
    int wave = tid >> 5, lane = tid & 31;
    int mi = (wave >> 2) << 4;   // 0 or 16
    int hi = (wave & 3) << 4;    // 0,16,32,48
    v8f acc = {};
#pragma unroll
    for (int kc = 0; kc < 4; ++kc) {
        v16bf a  = gather_frag(ldsA, mi, kc * 32, lane);
        v16bf bb = gather_frag(ldsB, hi, kc * 32, lane);
        acc = __builtin_amdgcn_wmma_f32_16x16x32_bf16(false, a, false, bb,
                                                      (short)0, acc, false, false);
    }
    int   h  = hi + (lane & 15);
    float vh = vec[h];
    float bh = bias[h];
    int mbase = mi + ((lane >> 4) << 3);
#pragma unroll
    for (int r = 0; r < 8; ++r)
        atomicAdd(&scoreAcc[mbase + r], (acc[r] + bh) * vh);
}

// ---------------------------------------------------------------------------
// emb[b,e] = x[b,:] . I_w[e,:] + I_b[e]
// ---------------------------------------------------------------------------
__global__ void embed_kernel(const float* __restrict__ x, const float* __restrict__ I_w,
                             const float* __restrict__ I_b, float* __restrict__ emb) {
    int idx = blockIdx.x * 256 + threadIdx.x;
    if (idx >= B_ * E_) return;
    int b = idx / E_, e = idx % E_;
    const float* xv = x + (size_t)b * IN_;
    const float* wv = I_w + (size_t)e * IN_;
    float acc = I_b[e];
    for (int i = 0; i < IN_; ++i) acc += xv[i] * wv[i];
    emb[idx] = acc;
}

// ---------------------------------------------------------------------------
// Wi[b,h] = emb[b,:] . W1_w[h,:] + W1_b[h];  Rq[b,h] = query[b,:] . R1_w[h,:] + R1_b[h]
// ---------------------------------------------------------------------------
__global__ void wi_rq_kernel(const float* __restrict__ emb, const float* __restrict__ query,
                             const float* __restrict__ W1_w, const float* __restrict__ W1_b,
                             const float* __restrict__ R1_w, const float* __restrict__ R1_b,
                             float* __restrict__ Wi, float* __restrict__ Rq) {
    int idx = blockIdx.x * 256 + threadIdx.x;
    if (idx >= B_ * H_) return;
    int b = idx / H_, h = idx % H_;
    const float* ev = emb + (size_t)b * E_;
    const float* w1 = W1_w + (size_t)h * E_;
    float a1 = W1_b[h];
    for (int e = 0; e < E_; ++e) a1 += ev[e] * w1[e];
    Wi[idx] = a1;
    const float* qv = query + (size_t)b * Q_;
    const float* r1 = R1_w + (size_t)h * Q_;
    float a2 = R1_b[h];
    for (int q = 0; q < Q_; ++q) a2 += qv[q] * r1[q];
    Rq[idx] = a2;
}

// ---------------------------------------------------------------------------
// Allocation weights: usage update, bitonic argsort of u over M in LDS,
// inclusive cumprod via segmented product scan, scatter back.
// ---------------------------------------------------------------------------
__global__ void alloc_kernel(const float* __restrict__ readw, const float* __restrict__ writew,
                             const float* __restrict__ usage, const float* __restrict__ freeg,
                             float* __restrict__ new_usage_out, float* __restrict__ alloc_out) {
    int b = blockIdx.x, tid = threadIdx.x;
    __shared__ float          vals[M_];
    __shared__ unsigned short idxs[M_];
    __shared__ float          segred[256];
    size_t base = (size_t)b * M_;

    for (int m = tid; m < M_; m += 256) {
        float u   = 1e-5f + (1.0f - 1e-5f) * usage[base + m];
        float psi = 1.0f - freeg[base + m] * readw[base + m];
        float ww  = writew[base + m];
        u = u + ww - u * ww;
        u = u * psi;
        new_usage_out[base + m] = u;
        vals[m] = u;
        idxs[m] = (unsigned short)m;
    }
    __syncthreads();

    // bitonic sort ascending (2048 elements, 256 threads)
    for (int k = 2; k <= M_; k <<= 1) {
        for (int j = k >> 1; j > 0; j >>= 1) {
            for (int i = tid; i < M_; i += 256) {
                int ixj = i ^ j;
                if (ixj > i) {
                    bool up = ((i & k) == 0);
                    float vi = vals[i], vj = vals[ixj];
                    if ((vi > vj) == up) {
                        vals[i] = vj; vals[ixj] = vi;
                        unsigned short t = idxs[i]; idxs[i] = idxs[ixj]; idxs[ixj] = t;
                    }
                }
            }
            __syncthreads();
        }
    }

    // inclusive product scan: thread t owns [t*8, t*8+8)
    float loc[8];
    float prod = 1.0f;
    int s0 = tid * 8;
#pragma unroll
    for (int i = 0; i < 8; ++i) { prod *= vals[s0 + i]; loc[i] = prod; }
    segred[tid] = prod;
    __syncthreads();
    for (int off = 1; off < 256; off <<= 1) {
        float v = segred[tid];
        float p = (tid >= off) ? segred[tid - off] : 1.0f;
        __syncthreads();
        segred[tid] = v * p;
        __syncthreads();
    }
    float prefix = (tid == 0) ? 1.0f : segred[tid - 1];

#pragma unroll
    for (int i = 0; i < 8; ++i) {
        int g = s0 + i;
        float cum = prefix * loc[i];
        alloc_out[base + idxs[g]] = (1.0f - vals[g]) * cum;
    }
}

// ---------------------------------------------------------------------------
// Fused Wm = memory @ W2^T + W2_b  then  scores[m] = Wm[m,:] . Wi[b,:]
// A tile and weights staged via async global->LDS (ASYNCcnt).
// ---------------------------------------------------------------------------
__global__ void __launch_bounds__(256) wm_scores_kernel(
        const float* __restrict__ memory, const float* __restrict__ W2_w,
        const float* __restrict__ W2_b, const float* __restrict__ Wi,
        float* __restrict__ scores) {
    int m0 = blockIdx.x * 32;
    int b  = blockIdx.y;
    int tid = threadIdx.x;
    __shared__ float ldsA[32 * E_];   // 16 KB
    __shared__ float ldsB[H_ * E_];   // 32 KB
    __shared__ float vec[H_], bias[H_], scoreAcc[32];

    size_t gbase = ((size_t)b * M_ + m0) * E_;
    // async stage: A tile (1024 x b128) + B weights (2048 x b128)
    for (int t = tid; t < (32 * E_) / 4; t += 256)
        async_copy_b128(lds_off_of(&ldsA[t * 4]), memory + gbase + (size_t)t * 4);
    for (int t = tid; t < (H_ * E_) / 4; t += 256)
        async_copy_b128(lds_off_of(&ldsB[t * 4]), W2_w + (size_t)t * 4);
    if (tid < H_) { vec[tid] = Wi[(size_t)b * H_ + tid]; bias[tid] = W2_b[tid]; }
    if (tid < 32) scoreAcc[tid] = 0.f;
    wait_async_all();
    __syncthreads();

    wmma_tile_scores(ldsA, ldsB, vec, bias, scoreAcc, tid);
    __syncthreads();

    if (tid < 32) scores[(size_t)b * M_ + m0 + tid] = scoreAcc[tid];
}

// ---------------------------------------------------------------------------
// Softmax over M per batch; optionally combine with alloc: out = (p + alloc)/2
// ---------------------------------------------------------------------------
__global__ void softmax_combine_kernel(const float* __restrict__ scores,
                                       const float* __restrict__ alloc,
                                       float* __restrict__ out) {
    int b = blockIdx.x, tid = threadIdx.x;
    __shared__ float p[M_];
    __shared__ float red[256];
    size_t base = (size_t)b * M_;

    float mx = -3.402823466e38f;
    for (int m = tid; m < M_; m += 256) { float v = scores[base + m]; p[m] = v; mx = fmaxf(mx, v); }
    red[tid] = mx;
    __syncthreads();
    for (int off = 128; off > 0; off >>= 1) {
        if (tid < off) red[tid] = fmaxf(red[tid], red[tid + off]);
        __syncthreads();
    }
    mx = red[0];
    __syncthreads();

    float s = 0.f;
    for (int m = tid; m < M_; m += 256) { float e = __expf(p[m] - mx); p[m] = e; s += e; }
    red[tid] = s;
    __syncthreads();
    for (int off = 128; off > 0; off >>= 1) {
        if (tid < off) red[tid] += red[tid + off];
        __syncthreads();
    }
    float inv = 1.0f / red[0];

    if (alloc) {
        for (int m = tid; m < M_; m += 256)
            out[base + m] = (p[m] * inv + alloc[base + m]) * 0.5f;
    } else {
        for (int m = tid; m < M_; m += 256)
            out[base + m] = p[m] * inv;
    }
}

// ---------------------------------------------------------------------------
// Fused: new_memory = memory + w^T*emb (update applied in place in LDS,
// f32 written to HBM once), then Rm = new_memory @ R2^T + R2_b and
// rscores[m] = Rm[m,:] . Rq[b,:] straight out of the updated LDS tile.
// ---------------------------------------------------------------------------
__global__ void __launch_bounds__(256) update_rm_kernel(
        const float* __restrict__ memory, const float* __restrict__ w,
        const float* __restrict__ emb, const float* __restrict__ R2_w,
        const float* __restrict__ R2_b, const float* __restrict__ Rq,
        float* __restrict__ newmem_out, float* __restrict__ rscores) {
    int m0 = blockIdx.x * 32;
    int b  = blockIdx.y;
    int tid = threadIdx.x;
    __shared__ float ldsA[32 * E_];   // 16 KB, updated in place
    __shared__ float ldsB[H_ * E_];   // 32 KB
    __shared__ float vec[H_], bias[H_], scoreAcc[32];
    __shared__ float wrow[32], embv[E_];

    size_t base_m = (size_t)b * M_ + m0;
    size_t gbase  = base_m * E_;

    for (int t = tid; t < (32 * E_) / 4; t += 256)
        async_copy_b128(lds_off_of(&ldsA[t * 4]), memory + gbase + (size_t)t * 4);
    for (int t = tid; t < (H_ * E_) / 4; t += 256)
        async_copy_b128(lds_off_of(&ldsB[t * 4]), R2_w + (size_t)t * 4);
    if (tid < H_) { vec[tid] = Rq[(size_t)b * H_ + tid]; bias[tid] = R2_b[tid]; }
    if (tid < 32) { wrow[tid] = w[base_m + tid]; scoreAcc[tid] = 0.f; }
    if (tid < E_) embv[tid] = emb[(size_t)b * E_ + tid];
    wait_async_all();
    __syncthreads();

    // rank-1 update in place; single f32 store of new_memory to HBM
    for (int t = tid; t < 32 * E_; t += 256) {
        int r = t >> 7, e = t & (E_ - 1);
        float nm = ldsA[t] + wrow[r] * embv[e];
        newmem_out[gbase + t] = nm;
        ldsA[t] = nm;
    }
    __syncthreads();

    wmma_tile_scores(ldsA, ldsB, vec, bias, scoreAcc, tid);
    __syncthreads();

    if (tid < 32) rscores[base_m + tid] = scoreAcc[tid];
}

// ---------------------------------------------------------------------------
// pooled[b,e] = sum_m wr[b,m] * new_memory[b,m,e]
// ---------------------------------------------------------------------------
__global__ void pooled_kernel(const float* __restrict__ wr, const float* __restrict__ newmem,
                              float* __restrict__ pooled) {
    int b = blockIdx.x, tid = threadIdx.x;       // 512 threads
    int e = tid & (E_ - 1), mg = tid >> 7;       // 4 m-groups
    __shared__ float part[4][E_];
    size_t mb = (size_t)b * M_;
    size_t nb = (size_t)b * M_ * E_;
    float acc = 0.f;
    for (int m = mg; m < M_; m += 4) acc += wr[mb + m] * newmem[nb + (size_t)m * E_ + e];
    part[mg][e] = acc;
    __syncthreads();
    if (mg == 0) pooled[(size_t)b * E_ + e] = part[0][e] + part[1][e] + part[2][e] + part[3][e];
}

// ---------------------------------------------------------------------------
// output[b,o] = pooled[b,:] . O_w[o,:] + O_b[o]
// ---------------------------------------------------------------------------
__global__ void output_kernel(const float* __restrict__ pooled, const float* __restrict__ O_w,
                              const float* __restrict__ O_b, float* __restrict__ out) {
    int idx = blockIdx.x * 256 + threadIdx.x;
    if (idx >= B_ * OUT_) return;
    int b = idx >> 9, o = idx & (OUT_ - 1);
    const float* p = pooled + (size_t)b * E_;
    const float* wv = O_w + (size_t)o * E_;
    float acc = O_b[o];
    for (int e = 0; e < E_; ++e) acc += p[e] * wv[e];
    out[idx] = acc;
}

// ---------------------------------------------------------------------------
extern "C" void kernel_launch(void* const* d_in, const int* in_sizes, int n_in,
                              void* d_out, int out_size, void* d_ws, size_t ws_size,
                              hipStream_t stream) {
    const float* read_w  = (const float*)d_in[0];
    const float* write_w = (const float*)d_in[1];
    const float* usage   = (const float*)d_in[2];
    const float* free_g  = (const float*)d_in[3];
    const float* x       = (const float*)d_in[4];
    const float* query   = (const float*)d_in[5];
    const float* memory  = (const float*)d_in[6];
    const float* I_w  = (const float*)d_in[7];
    const float* I_b  = (const float*)d_in[8];
    const float* W1_w = (const float*)d_in[9];
    const float* W1_b = (const float*)d_in[10];
    const float* W2_w = (const float*)d_in[11];
    const float* W2_b = (const float*)d_in[12];
    const float* R1_w = (const float*)d_in[13];
    const float* R1_b = (const float*)d_in[14];
    const float* R2_w = (const float*)d_in[15];
    const float* R2_b = (const float*)d_in[16];
    const float* O_w  = (const float*)d_in[17];
    const float* O_b  = (const float*)d_in[18];

    float* out        = (float*)d_out;
    float* out_wr     = out;
    float* out_nww    = out + (size_t)B_ * M_;
    float* out_nusage = out + 2 * (size_t)B_ * M_;
    float* out_y      = out + 3 * (size_t)B_ * M_;
    float* out_newmem = out + 3 * (size_t)B_ * M_ + (size_t)B_ * OUT_;

    float* ws = (float*)d_ws;
    float* ws_emb    = ws; ws += (size_t)B_ * E_;
    float* ws_Wi     = ws; ws += (size_t)B_ * H_;
    float* ws_Rq     = ws; ws += (size_t)B_ * H_;
    float* ws_alloc  = ws; ws += (size_t)B_ * M_;
    float* ws_scores = ws; ws += (size_t)B_ * M_;
    float* ws_rsc    = ws; ws += (size_t)B_ * M_;
    float* ws_pooled = ws; ws += (size_t)B_ * E_;

    embed_kernel<<<(B_ * E_ + 255) / 256, 256, 0, stream>>>(x, I_w, I_b, ws_emb);
    wi_rq_kernel<<<(B_ * H_ + 255) / 256, 256, 0, stream>>>(ws_emb, query, W1_w, W1_b,
                                                            R1_w, R1_b, ws_Wi, ws_Rq);
    alloc_kernel<<<B_, 256, 0, stream>>>(read_w, write_w, usage, free_g,
                                         out_nusage, ws_alloc);
    wm_scores_kernel<<<dim3(M_ / 32, B_), 256, 0, stream>>>(memory, W2_w, W2_b,
                                                            ws_Wi, ws_scores);
    softmax_combine_kernel<<<B_, 256, 0, stream>>>(ws_scores, ws_alloc, out_nww);
    update_rm_kernel<<<dim3(M_ / 32, B_), 256, 0, stream>>>(memory, out_nww, ws_emb,
                                                            R2_w, R2_b, ws_Rq,
                                                            out_newmem, ws_rsc);
    softmax_combine_kernel<<<B_, 256, 0, stream>>>(ws_rsc, nullptr, out_wr);
    pooled_kernel<<<B_, 512, 0, stream>>>(out_wr, out_newmem, ws_pooled);
    output_kernel<<<(B_ * OUT_ + 255) / 256, 256, 0, stream>>>(ws_pooled, O_w, O_b, out_y);
}
